// MultiHeadSelfAttention_35433480192221
// MI455X (gfx1250) — matrix operand order repaired and checked by reading, not compiled, not run on hardware
//
#include <hip/hip_runtime.h>
#include <hip/hip_bf16.h>

// ---------------------------------------------------------------------------
// MI455X (gfx1250) multi-head self-attention, f16 WMMA pipeline.
// B=4, N=2048, C=768, H=12, D=64.  ~90 GFLOP vs ~150MB traffic -> compute
// bound -> everything on v_wmma_f32_16x16x32_f16 (f32 accumulate).
// Round 3: attention workgroups (8 waves) now share one (b,h) and stage K/V
// tiles into LDS with gfx1250 async copies (global_load_async_to_lds_b128,
// ASYNCcnt), double-buffered -> 8x less global/L2 read traffic and DMA/WMMA
// overlap.  GEMMs unchanged (round-2 codegen was spill-free and clean).
// ---------------------------------------------------------------------------

typedef __attribute__((ext_vector_type(16))) _Float16 v16h;
typedef __attribute__((ext_vector_type(8)))  _Float16 v8h;
typedef __attribute__((ext_vector_type(8)))  float    v8f;

#define WMMA16(a, b, c) \
    __builtin_amdgcn_wmma_f32_16x16x32_f16(false, (a), false, (b), (short)0, (c), false, false)

#define B_      4
#define N_      2048
#define C_      768
#define H_      12
#define D_      64
#define NTOK    (B_ * N_)        // 8192
#define NQKV    (3 * C_)         // 2304
#define SCALE_  0.125f           // 1/sqrt(64)

static __device__ __forceinline__ v16h cat16(v8h lo, v8h hi) {
    v16h r;
#pragma unroll
    for (int i = 0; i < 8; ++i) { r[i] = lo[i]; r[i + 8] = hi[i]; }
    return r;
}

// A-fragment (16x32 f16): lane<16 holds row m, K = kk..kk+15;
// lane>=16 holds row m, K = kk+16..kk+31.  p = row base + kk.
static __device__ __forceinline__ v16h ld_afrag(const _Float16* p, bool hiHalf) {
    return *(const v16h*)(p + (hiHalf ? 16 : 0));
}

// Low 32 bits of a generic pointer to a __shared__ object = LDS byte offset
static __device__ __forceinline__ unsigned lds_off(const void* p) {
    return (unsigned)(uintptr_t)p;
}

// Async DMA: copy 16 bytes per lane from global memory to LDS (ASYNCcnt).
static __device__ __forceinline__ void async_cp16(unsigned lds, const void* gp) {
    asm volatile("global_load_async_to_lds_b128 %0, %1, off"
                 :: "v"(lds), "v"((unsigned long long)(uintptr_t)gp)
                 : "memory");
}
static __device__ __forceinline__ void wait_async0() {
    asm volatile("s_wait_asynccnt 0" ::: "memory");
}

// ---------------------------------------------------------------------------
// Conversion kernels
// ---------------------------------------------------------------------------
__global__ void cvt_f32_to_f16(_Float16* __restrict__ dst, const float* __restrict__ src, int n) {
    int i = blockIdx.x * blockDim.x + threadIdx.x;
    if (i < n) dst[i] = (_Float16)src[i];
}

// dst[NROWS][K] = (f16) src[K][NROWS]   (transpose so WMMA B-frags are contiguous)
__global__ void cvt_transpose_f16(_Float16* __restrict__ dst, const float* __restrict__ src,
                                  int nrows, int k) {
    int i = blockIdx.x * blockDim.x + threadIdx.x;
    if (i < nrows * k) {
        int rn = i / k, rk = i - rn * k;
        dst[i] = (_Float16)src[(size_t)rk * nrows + rn];
    }
}

// ---------------------------------------------------------------------------
// QKV projection: [8192,768] x [768,2304] + bias  ->  Q[b,h,n,d] K[b,h,n,d]
// Vt[b,h,d,n] (all f16).  One wave computes a 64x32 output tile (8 acc frags
// = 64 VGPRs; total live state ~112 VGPRs -> no spills).
// ---------------------------------------------------------------------------
__global__ void __launch_bounds__(256, 1)
qkv_gemm(const _Float16* __restrict__ Xh, const _Float16* __restrict__ Wt,
         const float* __restrict__ bias,
         _Float16* __restrict__ Qh, _Float16* __restrict__ Kh,
         _Float16* __restrict__ Vt) {
    const int lane = threadIdx.x & 31;
    const int wid  = threadIdx.x >> 5;
    const int task = blockIdx.x * 8 + wid;            // 128 * 72 = 9216 tiles
    if (task >= (NTOK / 64) * (NQKV / 32)) return;
    const int tm = task / (NQKV / 32), tn = task % (NQKV / 32);
    const int row0 = tm * 64, col0 = tn * 32;
    const int lo16 = lane & 15;
    const bool hiHalf = lane >= 16;

    v8f acc[4][2] = {};
#pragma unroll 1
    for (int kk = 0; kk < C_; kk += 32) {
        v16h a[4], b[2];
#pragma unroll
        for (int im = 0; im < 4; ++im)
            a[im] = ld_afrag(Xh + (size_t)(row0 + im * 16 + lo16) * C_ + kk, hiHalf);
#pragma unroll
        for (int in = 0; in < 2; ++in)
            b[in] = *(const v16h*)(Wt + (size_t)(col0 + in * 16 + lo16) * C_ + kk + (hiHalf ? 16 : 0));
#pragma unroll
        for (int im = 0; im < 4; ++im)
#pragma unroll
            for (int in = 0; in < 2; ++in)
                acc[im][in] = WMMA16(a[im], b[in], acc[im][in]);
    }

#pragma unroll
    for (int im = 0; im < 4; ++im) {
#pragma unroll
        for (int in = 0; in < 2; ++in) {
            const int col   = col0 + in * 16 + lo16;        // 0..2303
            const float bc  = bias[col];
            const int which = col / C_;                     // 0=Q 1=K 2=V (uniform per frag)
            const int rem   = col - which * C_;
            const int h = rem >> 6, d = rem & 63;
            const int rm = row0 + im * 16 + (hiHalf ? 8 : 0);
            const int bb = rm >> 11, n0 = rm & (N_ - 1);    // no 2048-crossing inside 8 rows
            if (which == 2) {
                v8h o;
#pragma unroll
                for (int i = 0; i < 8; ++i) o[i] = (_Float16)(acc[im][in][i] + bc);
                *(v8h*)(Vt + (((size_t)(bb * H_ + h)) * D_ + d) * N_ + n0) = o;
            } else {
                _Float16* dst = (which == 0 ? Qh : Kh) +
                                (((size_t)(bb * H_ + h)) * N_ + n0) * D_ + d;
#pragma unroll
                for (int i = 0; i < 8; ++i) dst[(size_t)i * D_] = (_Float16)(acc[im][in][i] + bc);
            }
        }
    }
}

// ---------------------------------------------------------------------------
// Flash attention.  One block (8 waves) owns 128 queries of one (b,h); the
// block cooperatively double-buffers 32-key K/V tiles in LDS via async DMA.
// Per wave: S^T = K * Q^T (each lane owns one query column -> softmax is
// intra-lane + shfl_xor(16)); O^T = V^T * P^T with P^T re-laid-out through
// per-wave LDS.
// ---------------------------------------------------------------------------
__global__ void __launch_bounds__(256)
attn_kernel(const _Float16* __restrict__ Qh, const _Float16* __restrict__ Kh,
            const _Float16* __restrict__ Vt, _Float16* __restrict__ Oh) {
    __shared__ alignas(32) _Float16 Ks[2][32 * 64];   // [key][d]   4KB each
    __shared__ alignas(32) _Float16 Vs[2][64 * 32];   // [d][key]   4KB each
    __shared__ alignas(32) _Float16 Pt[8][16 * 32];   // per-wave P^T tile

    const int t    = threadIdx.x;
    const int lane = t & 31;
    const int wid  = t >> 5;
    const int bh   = blockIdx.x >> 4;                 // b*12 + h   (48 values)
    const int qch  = blockIdx.x & 15;                 // 16 chunks of 128 queries
    const int lo16 = lane & 15;
    const bool hiHalf = lane >= 16;
    const int pbase = hiHalf ? 8 : 0;
    const int q = qch * 128 + wid * 16 + lo16;        // query this lane owns

    const _Float16* Qb = Qh + (size_t)bh * N_ * D_;
    const _Float16* Kb = Kh + (size_t)bh * N_ * D_;
    const _Float16* Vb = Vt + (size_t)bh * D_ * N_;
    _Float16* myPt = Pt[wid];

    // Per-thread staging slots: K tile is one contiguous 4KB region (32 keys x
    // 64 d); V tile is 64 rows (d) x 64B (32 keys).  16B per thread per tile.
    const _Float16* gK = Kb + t * 8;                          // + j*64 per step
    const _Float16* gV = Vb + (size_t)(t >> 2) * N_ + (t & 3) * 8;  // + j per step
    const unsigned lK[2] = { lds_off(&Ks[0][t * 8]), lds_off(&Ks[1][t * 8]) };
    const unsigned lV[2] = { lds_off(&Vs[0][(t >> 2) * 32 + (t & 3) * 8]),
                             lds_off(&Vs[1][(t >> 2) * 32 + (t & 3) * 8]) };

    // Q^T B-fragments: lane = query column, 16 contiguous d values.
    v16h qf[2];
    qf[0] = *(const v16h*)(Qb + (size_t)q * D_ + 0  + (hiHalf ? 16 : 0));
    qf[1] = *(const v16h*)(Qb + (size_t)q * D_ + 32 + (hiHalf ? 16 : 0));

    float m = -3.0e38f, l = 0.0f;
    v8f acc[4] = {};

    // Prologue: stage tile j=0 into buffer 0.
    async_cp16(lK[0], gK);
    async_cp16(lV[0], gV);
    wait_async0();
    __syncthreads();

    int buf = 0;
#pragma unroll 1
    for (int j = 0; j < N_; j += 32) {
        if (j + 32 < N_) {            // async-stage next tile into buf^1
            async_cp16(lK[buf ^ 1], gK + (size_t)(j + 32) * D_);
            async_cp16(lV[buf ^ 1], gV + (j + 32));
        }
        const _Float16* Kt = Ks[buf];
        const _Float16* Vtile = Vs[buf];

        // S^T tiles: keys 0..15 and 16..31 of this LDS tile (rows) x queries
        v8f s0 = {}, s1 = {};
#pragma unroll
        for (int kk2 = 0; kk2 < 2; ++kk2) {
            v16h a0 = ld_afrag(Kt + (size_t)lo16 * D_ + kk2 * 32, hiHalf);
            v16h a1 = ld_afrag(Kt + (size_t)(16 + lo16) * D_ + kk2 * 32, hiHalf);
            s0 = WMMA16(a0, qf[kk2], s0);
            s1 = WMMA16(a1, qf[kk2], s1);
        }
        // online softmax over the 32 new keys of this lane's query
        float mx = s0[0];
#pragma unroll
        for (int i = 0; i < 8; ++i) { mx = fmaxf(mx, s0[i]); mx = fmaxf(mx, s1[i]); }
        mx *= SCALE_;
        mx = fmaxf(mx, __shfl_xor(mx, 16, 32));      // combine the two key halves
        const float mn = fmaxf(m, mx);
        const float alpha = __expf(m - mn);
        l *= alpha;
#pragma unroll
        for (int dt = 0; dt < 4; ++dt) acc[dt] *= alpha;
        v8h p0v, p1v;
#pragma unroll
        for (int i = 0; i < 8; ++i) {
            float p0 = __expf(s0[i] * SCALE_ - mn); l += p0; p0v[i] = (_Float16)p0;
            float p1 = __expf(s1[i] * SCALE_ - mn); l += p1; p1v[i] = (_Float16)p1;
        }
        m = mn;
        *(v8h*)(myPt + lo16 * 32 + pbase)      = p0v;   // keys base..base+7
        *(v8h*)(myPt + lo16 * 32 + 16 + pbase) = p1v;   // keys 16+base..+7
        // B-fragment of P^T (32 keys x 16 queries), same-wave LDS -> in order
        v16h pb = *(const v16h*)(myPt + lo16 * 32 + (hiHalf ? 16 : 0));
        // O^T += V^T * P^T  (rows = d, cols = queries)
#pragma unroll
        for (int dt = 0; dt < 4; ++dt) {
            v16h vf = ld_afrag(Vtile + (size_t)(dt * 16 + lo16) * 32, hiHalf);
            acc[dt] = WMMA16(vf, pb, acc[dt]);
        }

        wait_async0();      // our copies into buf^1 are complete
        __syncthreads();    // everyone done reading buf & done staging buf^1
        buf ^= 1;
    }

    const float lt  = l + __shfl_xor(l, 16, 32);
    const float inv = 1.0f / lt;
    const int b = bh / H_, h = bh - b * H_;
    const size_t rowOff = ((size_t)(b * N_ + q)) * C_ + h * D_;
#pragma unroll
    for (int dt = 0; dt < 4; ++dt) {
        v8h o;
#pragma unroll
        for (int i = 0; i < 8; ++i) o[i] = (_Float16)(acc[dt][i] * inv);
        *(v8h*)(Oh + rowOff + dt * 16 + pbase) = o;
    }
}

// ---------------------------------------------------------------------------
// Output projection: [8192,768] x [768,768] + bias -> f32 out.
// 64x32 wave tile, same structure as qkv_gemm.
// ---------------------------------------------------------------------------
__global__ void __launch_bounds__(256, 1)
proj_gemm(const _Float16* __restrict__ Ah, const _Float16* __restrict__ Wt,
          const float* __restrict__ bias, float* __restrict__ out) {
    const int lane = threadIdx.x & 31;
    const int wid  = threadIdx.x >> 5;
    const int task = blockIdx.x * 8 + wid;            // 128 * 24 = 3072 tiles
    if (task >= (NTOK / 64) * (C_ / 32)) return;
    const int tm = task / (C_ / 32), tn = task % (C_ / 32);
    const int row0 = tm * 64, col0 = tn * 32;
    const int lo16 = lane & 15;
    const bool hiHalf = lane >= 16;

    v8f acc[4][2] = {};
#pragma unroll 1
    for (int kk = 0; kk < C_; kk += 32) {
        v16h a[4], b[2];
#pragma unroll
        for (int im = 0; im < 4; ++im)
            a[im] = ld_afrag(Ah + (size_t)(row0 + im * 16 + lo16) * C_ + kk, hiHalf);
#pragma unroll
        for (int in = 0; in < 2; ++in)
            b[in] = *(const v16h*)(Wt + (size_t)(col0 + in * 16 + lo16) * C_ + kk + (hiHalf ? 16 : 0));
#pragma unroll
        for (int im = 0; im < 4; ++im)
#pragma unroll
            for (int in = 0; in < 2; ++in)
                acc[im][in] = WMMA16(a[im], b[in], acc[im][in]);
    }
#pragma unroll
    for (int im = 0; im < 4; ++im) {
#pragma unroll
        for (int in = 0; in < 2; ++in) {
            const int col = col0 + in * 16 + lo16;
            const float bc = bias[col];
            const int rm = row0 + im * 16 + (hiHalf ? 8 : 0);
            float* dst = out + (size_t)rm * C_ + col;
#pragma unroll
            for (int i = 0; i < 8; ++i) dst[(size_t)i * C_] = acc[im][in][i] + bc;
        }
    }
}

// ---------------------------------------------------------------------------
extern "C" void kernel_launch(void* const* d_in, const int* in_sizes, int n_in,
                              void* d_out, int out_size, void* d_ws, size_t ws_size,
                              hipStream_t stream) {
    const float* x      = (const float*)d_in[0];
    const float* qkv_w  = (const float*)d_in[1];
    const float* qkv_b  = (const float*)d_in[2];
    const float* proj_w = (const float*)d_in[3];
    const float* proj_b = (const float*)d_in[4];
    float* out = (float*)d_out;

    // workspace layout (f16 tensors, 256B aligned)
    char* w = (char*)d_ws;
    size_t off = 0;
    auto take = [&](size_t bytes) { char* p = w + off; off = (off + bytes + 255) & ~(size_t)255; return p; };
    _Float16* Xh = (_Float16*)take((size_t)NTOK * C_ * 2);       // x in f16
    _Float16* Wq = (_Float16*)take((size_t)NQKV * C_ * 2);       // qkv_w^T  [2304][768]
    _Float16* Pw = (_Float16*)take((size_t)C_ * C_ * 2);         // proj_w^T [768][768]
    _Float16* Qh = (_Float16*)take((size_t)B_ * H_ * N_ * D_ * 2);
    _Float16* Kh = (_Float16*)take((size_t)B_ * H_ * N_ * D_ * 2);
    _Float16* Vt = (_Float16*)take((size_t)B_ * H_ * D_ * N_ * 2);
    _Float16* Oh = (_Float16*)take((size_t)NTOK * C_ * 2);       // attention out, f16

    const int nx = NTOK * C_;              // 6,291,456
    cvt_f32_to_f16<<<(nx + 255) / 256, 256, 0, stream>>>(Xh, x, nx);
    const int nw = NQKV * C_;              // 1,769,472
    cvt_transpose_f16<<<(nw + 255) / 256, 256, 0, stream>>>(Wq, qkv_w, NQKV, C_);
    const int np = C_ * C_;                // 589,824
    cvt_transpose_f16<<<(np + 255) / 256, 256, 0, stream>>>(Pw, proj_w, C_, C_);

    qkv_gemm<<<(NTOK / 64) * (NQKV / 32) / 8, 256, 0, stream>>>(Xh, Wq, qkv_b, Qh, Kh, Vt);
    attn_kernel<<<B_ * H_ * (N_ / 128), 256, 0, stream>>>(Qh, Kh, Vt, Oh);
    proj_gemm<<<(NTOK / 64) * (C_ / 32) / 8, 256, 0, stream>>>(Oh, Pw, proj_b, out);
}
